// InfoGraph_64295660421905
// MI455X (gfx1250) — matrix operand order repaired and checked by reading, not compiled, not run on hardware
//
#include <hip/hip_runtime.h>
#include <hip/hip_bf16.h>

typedef __attribute__((ext_vector_type(16))) _Float16 v16h;
typedef __attribute__((ext_vector_type(8)))  _Float16 v8h;
typedef __attribute__((ext_vector_type(8)))  float    v8f;
typedef __attribute__((ext_vector_type(4)))  float    v4f;

#define N_NODES   100000
#define IN_DIM    2048
#define HID       512
#define OUT_DIM   128
#define N_GRAPHS  512
#define MPAD      100032            // 1563 * 64
#define LOG2C     0.6931471805599453f

// ---------------- workspace layout (bytes, 256B aligned) ----------------
#define O_W1JT  0ul                                  // 640*2048 f16 = 2,621,440
#define O_W2T   2621440ul                            // 512*512  f16 =   524,288
#define O_W3T   3145728ul                            // 128*512  f16 =   131,072
#define O_GH    3276800ul                            // 512*128  f16 =   131,072
#define O_ACC   3407872ul                            // 2 floats (padded to 256)
#define O_H1    3408128ul                            // MPAD*512 f16 = 102,432,768
#define O_H2    105840896ul                          // MPAD*512 f16
#define O_LENC  208273664ul                          // MPAD*128 f16 =  25,608,192
#define O_JUMP  233881856ul                          // MPAD*128 f32 =  51,216,384
#define WS_NEEDED 285098240ul

// ---------------- fragment loaders (CDNA5 wave32 WMMA layouts) ----------------
// A 16x32 f16 per lane: 16 contiguous halfwords (base pre-offset by hi*16)
__device__ __forceinline__ v16h load_a16(const _Float16* p) {
  union U { v16h v; v8h h[2]; } u;
  u.h[0] = *(const v8h*)p;
  u.h[1] = *(const v8h*)(p + 8);
  return u.v;
}
// Same fragment from an f32 source (streamed once -> non-temporal loads), cvt in-register
__device__ __forceinline__ v16h load_a32(const float* p) {
  union U { v16h v; v8h h[2]; } u;
#pragma unroll
  for (int c2 = 0; c2 < 2; ++c2) {
    v4f f0 = __builtin_nontemporal_load((const v4f*)(p + c2 * 8));
    v4f f1 = __builtin_nontemporal_load((const v4f*)(p + c2 * 8 + 4));
    v8h hh;
#pragma unroll
    for (int t = 0; t < 4; ++t) { hh[t] = (_Float16)f0[t]; hh[t + 4] = (_Float16)f1[t]; }
    u.h[c2] = hh;
  }
  return u.v;
}

// Ping-pong software-pipelined K loop (K must be a multiple of 64).
// Fragments are reloaded in place right after their WMMA chain -> no register rotation.
template <bool AF32, int NC>
__device__ __forceinline__ void wmma_kloop(const void* A, int lda, int arow, int hi,
                                           const _Float16* Bt, int ldb, int col0,
                                           int lane15, int K, v8f* acc) {
  const float*    ap32 = (const float*)A + (long)arow * lda + hi * 16;
  const _Float16* ap16 = (const _Float16*)A + (long)arow * lda + hi * 16;
  const _Float16* bp[NC];
#pragma unroll
  for (int c = 0; c < NC; ++c)
    bp[c] = Bt + (long)(col0 + c * 16 + lane15) * ldb + hi * 16;

  auto loadA = [&](int kk) -> v16h {
    if constexpr (AF32) return load_a32(ap32 + kk); else return load_a16(ap16 + kk);
  };
  auto loadB = [&](v16h* bb, int kk) {
#pragma unroll
    for (int c = 0; c < NC; ++c) bb[c] = *(const v16h*)(bp[c] + kk);
  };
  auto mma = [&](const v16h& aa, const v16h* bb) {
#pragma unroll
    for (int c = 0; c < NC; ++c)
      acc[c] = __builtin_amdgcn_wmma_f32_16x16x32_f16(false, aa, false, bb[c], (short)0,
                                                      acc[c], false, false);
  };

  v16h a0, a1, b0[NC], b1[NC];
  a0 = loadA(0);  loadB(b0, 0);
  a1 = loadA(32); loadB(b1, 32);
  int k = 0;
  for (; k + 64 < K; k += 64) {
    mma(a0, b0);
    a0 = loadA(k + 64); loadB(b0, k + 64);
    mma(a1, b1);
    a1 = loadA(k + 96); loadB(b1, k + 96);
  }
  mma(a0, b0);
  mma(a1, b1);
}

__device__ __forceinline__ float softplus_fast(float z) {
  return fmaxf(z, 0.f) + __logf(1.f + __expf(-fabsf(z)));  // v_exp_f32 / v_log_f32
}

// ---------------- kernel 0: weight transpose/convert + accum init ----------------
__global__ void cvt_weights_kernel(const float* W1, const float* W2, const float* W3,
                                   const float* Wj, const float* g_enc,
                                   _Float16* W1jt, _Float16* W2t, _Float16* W3t,
                                   _Float16* gh, float* accum) {
  int idx = blockIdx.x * blockDim.x + threadIdx.x;
  if (idx == 0) { accum[0] = 0.f; accum[1] = 0.f; }
  const int total = 640 * 2048 + 512 * 512 + 128 * 512 + 512 * 128;
  for (int i = idx; i < total; i += gridDim.x * blockDim.x) {
    int j = i;
    if (j < 640 * 2048) {
      int n = j / 2048, k = j % 2048;
      float v = (n < 512) ? W1[(long)k * 512 + n] : Wj[(long)k * 128 + (n - 512)];
      W1jt[j] = (_Float16)v;
    } else if ((j -= 640 * 2048) < 512 * 512) {
      int n = j / 512, k = j % 512;
      W2t[j] = (_Float16)W2[(long)k * 512 + n];
    } else if ((j -= 512 * 512) < 128 * 512) {
      int n = j / 512, k = j % 512;
      W3t[j] = (_Float16)W3[(long)k * 128 + n];
    } else {
      j -= 128 * 512;
      gh[j] = (_Float16)g_enc[j];                    // [512][128] K-contiguous = B layout
    }
  }
}

// ---------------- kernel 1: fused feat@[W1|Wj] ----------------
// 256 thr (8 waves). Block covers 64 rows x all 640 cols (feat streamed from HBM once).
__global__ void __launch_bounds__(256)
gemm1_kernel(const float* __restrict__ feat,
             const float* __restrict__ b1, const float* __restrict__ bj,
             const _Float16* __restrict__ W1jt,
             _Float16* __restrict__ h1, float* __restrict__ jump) {
  const int lane = threadIdx.x & 31, wave = threadIdx.x >> 5;
  const int lane15 = lane & 15, hi = lane >> 4;
  const int m0 = blockIdx.x * 64 + (wave & 3) * 16;
  const int lrow = m0 + lane15;
  const int arow = lrow < N_NODES ? lrow : N_NODES - 1;   // clamp: padded rows masked later
#pragma unroll
  for (int i = 0; i < 5; ++i) {
    const int col0 = ((wave >> 2) + 2 * i) * 64;          // column groups 0..9
    v8f acc[4] = {};
    wmma_kloop<true, 4>(feat, IN_DIM, arow, hi, W1jt, IN_DIM, col0, lane15, IN_DIM, acc);
#pragma unroll
    for (int c = 0; c < 4; ++c) {
      const int col = col0 + c * 16 + lane15;
#pragma unroll
      for (int v = 0; v < 8; ++v) {
        const int r = m0 + v + hi * 8;
        float x = acc[c][v];
        if (col < HID) {
          float y = x + b1[col];
          h1[(long)r * HID + col] = (_Float16)(y > 0.f ? y : 0.f);
        } else {
          jump[(long)r * OUT_DIM + (col - HID)] = x + bj[col - HID];
        }
      }
    }
  }
}

// ---------------- kernels 2/3: f16 GEMM; MODE 0 = relu(x+b), MODE 1 = x+b+extra ----------------
template <int MODE, int NGROUPS, int KDIM>
__global__ void __launch_bounds__(256)
gemm_f16_kernel(const _Float16* __restrict__ A, int lda,
                const _Float16* __restrict__ Bt, int ldb,
                const float* __restrict__ bias,
                const float* __restrict__ extra,
                _Float16* __restrict__ out, int ldo) {
  const int lane = threadIdx.x & 31, wave = threadIdx.x >> 5;
  const int lane15 = lane & 15, hi = lane >> 4;
  const int m0 = blockIdx.x * 64 + (wave & 3) * 16;
#pragma unroll
  for (int i = 0; i < NGROUPS / 2; ++i) {
    const int col0 = ((wave >> 2) + 2 * i) * 64;
    v8f acc[4] = {};
    wmma_kloop<false, 4>(A, lda, m0 + lane15, hi, Bt, ldb, col0, lane15, KDIM, acc);
#pragma unroll
    for (int c = 0; c < 4; ++c) {
      const int col = col0 + c * 16 + lane15;
#pragma unroll
      for (int v = 0; v < 8; ++v) {
        const int r = m0 + v + hi * 8;
        float x = acc[c][v] + bias[col];
        if (MODE == 0)
          out[(long)r * ldo + col] = (_Float16)(x > 0.f ? x : 0.f);
        else
          out[(long)r * ldo + col] = (_Float16)(x + extra[(long)r * ldo + col]);
      }
    }
  }
}

// ---------------- kernel 4: res = l_enc @ g_enc^T fused with JS-loss reduction ----------------
__global__ void __launch_bounds__(256)
loss_kernel(const _Float16* __restrict__ lenc, const _Float16* __restrict__ gh,
            const int* __restrict__ graph_id, float* __restrict__ accum) {
  const int lane = threadIdx.x & 31, wave = threadIdx.x >> 5;
  const int lane15 = lane & 15, hi = lane >> 4;
  const int m0 = blockIdx.x * 64 + (wave & 3) * 16;
  int gids[8];
#pragma unroll
  for (int v = 0; v < 8; ++v) {
    const int r = m0 + v + hi * 8;
    gids[v] = (r < N_NODES) ? graph_id[r] : -1;
  }
  float pos = 0.f, neg = 0.f;
#pragma unroll
  for (int i = 0; i < 4; ++i) {
    const int col0 = ((wave >> 2) + 2 * i) * 64;
    v8f acc[4] = {};
    wmma_kloop<false, 4>(lenc, OUT_DIM, m0 + lane15, hi, gh, OUT_DIM, col0, lane15, OUT_DIM, acc);
#pragma unroll
    for (int c = 0; c < 4; ++c) {
      const int col = col0 + c * 16 + lane15;
#pragma unroll
      for (int v = 0; v < 8; ++v) {
        if (gids[v] >= 0) {
          const float x = acc[c][v];
          if (col == gids[v]) pos += LOG2C - softplus_fast(-x);   // E_pos term
          else                neg += softplus_fast(x) - LOG2C;    // softplus(-x)+x-log2
        }
      }
    }
  }
  __shared__ float sp, sn;
  if (threadIdx.x == 0) { sp = 0.f; sn = 0.f; }
  __syncthreads();
  atomicAdd(&sp, pos);                 // ds_add_f32
  atomicAdd(&sn, neg);
  __syncthreads();
  if (threadIdx.x == 0) { atomicAdd(&accum[0], sp); atomicAdd(&accum[1], sn); }
}

__global__ void finalize_kernel(const float* __restrict__ accum, float* __restrict__ out) {
  if (threadIdx.x == 0 && blockIdx.x == 0) {
    const float Epos = accum[0] / (float)N_NODES;
    const float Eneg = accum[1] / ((float)N_NODES * (float)(N_GRAPHS - 1));
    out[0] = Eneg - Epos;
  }
}

// ---------------- host side ----------------
extern "C" void kernel_launch(void* const* d_in, const int* in_sizes, int n_in,
                              void* d_out, int out_size, void* d_ws, size_t ws_size,
                              hipStream_t stream) {
  (void)in_sizes; (void)n_in; (void)out_size;
  if (ws_size < WS_NEEDED) return;

  const float* feat     = (const float*)d_in[0];
  const float* g_enc    = (const float*)d_in[1];
  const int*   graph_id = (const int*)d_in[2];
  const float* W1 = (const float*)d_in[3];
  const float* b1 = (const float*)d_in[4];
  const float* W2 = (const float*)d_in[5];
  const float* b2 = (const float*)d_in[6];
  const float* W3 = (const float*)d_in[7];
  const float* b3 = (const float*)d_in[8];
  const float* Wj = (const float*)d_in[9];
  const float* bj = (const float*)d_in[10];

  char* ws = (char*)d_ws;
  _Float16* W1jt = (_Float16*)(ws + O_W1JT);
  _Float16* W2t  = (_Float16*)(ws + O_W2T);
  _Float16* W3t  = (_Float16*)(ws + O_W3T);
  _Float16* gh   = (_Float16*)(ws + O_GH);
  float*    accv = (float*)   (ws + O_ACC);
  _Float16* h1   = (_Float16*)(ws + O_H1);
  _Float16* h2   = (_Float16*)(ws + O_H2);
  _Float16* lenc = (_Float16*)(ws + O_LENC);
  float*    jump = (float*)   (ws + O_JUMP);

  const int MBLK = MPAD / 64;   // 1563 row-blocks

  cvt_weights_kernel<<<2048, 256, 0, stream>>>(W1, W2, W3, Wj, g_enc,
                                               W1jt, W2t, W3t, gh, accv);

  // h1 = relu(feat@W1+b1), jump = feat@Wj+bj   (feat streamed once, NT loads)
  gemm1_kernel<<<MBLK, 256, 0, stream>>>(feat, b1, bj, W1jt, h1, jump);

  // h2 = relu(h1@W2 + b2)
  gemm_f16_kernel<0, 8, HID><<<MBLK, 256, 0, stream>>>(h1, HID, W2t, HID, b2, nullptr, h2, HID);

  // l_enc = h2@W3 + b3 + jump
  gemm_f16_kernel<1, 2, HID><<<MBLK, 256, 0, stream>>>(h2, HID, W3t, HID, b3, jump, lenc, OUT_DIM);

  // res = l_enc@g_enc^T fused with masked softplus loss reduction
  loss_kernel<<<MBLK, 256, 0, stream>>>(lenc, gh, graph_id, accv);

  finalize_kernel<<<1, 32, 0, stream>>>(accv, (float*)d_out);
}
